// Encoder_19430432047223
// MI455X (gfx1250) — compile-verified, matrix operand order checked by reading, not run on hardware
//
#include <hip/hip_runtime.h>
#include <hip/hip_bf16.h>

typedef __attribute__((ext_vector_type(16))) __bf16 v16bf;
typedef __attribute__((ext_vector_type(8)))  __bf16 v8bf;
typedef __attribute__((ext_vector_type(8)))  float  v8f;

#define EMB   256
#define UNITS 1024
#define BATCH 64
#define SEQ   256
#define G3    (3*UNITS)

// ---- fragment loaders -------------------------------------------------------
// A (16x32 bf16, row-major source): lane l holds row (l&15); elements 0..7 are
// cols kbase+0..7, elements 8..15 are cols kbase+16..23, kbase = 8*(l>>4).
static __device__ __forceinline__ v16bf load_a(const __bf16* p) {
  union { v16bf v; v8bf h[2]; } u;
  u.h[0] = *(const v8bf*)(p);        // 16B
  u.h[1] = *(const v8bf*)(p + 16);   // 16B
  return u.v;
}
// B (32x16 bf16, pre-packed): lane's 16 elements contiguous (32 bytes).
static __device__ __forceinline__ v16bf load_b(const __bf16* p) {
  union { v16bf v; v8bf h[2]; } u;
  u.h[0] = *(const v8bf*)(p);
  u.h[1] = *(const v8bf*)(p + 8);
  return u.v;
}
static __device__ __forceinline__ v8f wmma_bf16(v16bf a, v16bf b, v8f c) {
  return __builtin_amdgcn_wmma_f32_16x16x32_bf16(false, a, false, b, (short)0, c,
                                                 false, false);
}
static __device__ __forceinline__ float sigmoidf_(float v) {
  return 1.0f / (1.0f + __expf(-v));
}

// ---- setup kernels ----------------------------------------------------------
// Embedding gather + bf16 cast: xe[bt][e] = bf16(emb[x[bt]][e]); 4 elems/thread
__global__ void k_embed(const int* __restrict__ x, const float* __restrict__ emb,
                        __bf16* __restrict__ xe) {
  int i  = blockIdx.x * blockDim.x + threadIdx.x;   // BATCH*SEQ*EMB/4 threads
  int e4 = i & (EMB/4 - 1);
  int bt = i >> 6;
  int tok = x[bt];
  float4 f = *(const float4*)(emb + (size_t)tok * EMB + e4 * 4);
  __bf16* d = xe + (size_t)bt * EMB + e4 * 4;
  d[0] = (__bf16)f.x; d[1] = (__bf16)f.y; d[2] = (__bf16)f.z; d[3] = (__bf16)f.w;
}

// Pack fp32 (K x N row-major) weights into bf16 WMMA-B fragment block order:
// dst[((kt*NT + nt)*32 + lane)*16 + elem]
__global__ void k_pack(const float* __restrict__ src, __bf16* __restrict__ dst,
                       int K, int N) {
  int i = blockIdx.x * blockDim.x + threadIdx.x;
  if (i >= K * N) return;
  int k = i / N, n = i - k * N;
  int kt = k >> 5, kk = k & 31, nt = n >> 4, nn = n & 15;
  int lane = nn + 16 * ((kk >> 3) & 1);
  int elem = (kk & 7) + 8 * (kk >> 4);
  int NT = N >> 4;
  dst[(((size_t)kt * NT + nt) * 32 + lane) * 16 + elem] = (__bf16)src[i];
}

// Initial hidden (fp32) -> bf16 into both directions' ping buffers
__global__ void k_h0(const float* __restrict__ hidden, __bf16* __restrict__ hf,
                     __bf16* __restrict__ hb) {
  int i = blockIdx.x * blockDim.x + threadIdx.x;    // BATCH*UNITS
  __bf16 v = (__bf16)hidden[i];
  hf[i] = v; hb[i] = v;
}

// ---- one GRU time step, both directions -------------------------------------
// grid = 32 blocks x 256 threads. block>>4 = direction, block&15 = 64-unit slice.
// Each of 8 waves owns 2 (mt,nt) 16x16 tiles; accumulates Gz,Gr,Xh,Rh and applies
// the gate nonlinearity in-register.
__global__ __launch_bounds__(256) void k_gru_step(
    int s,
    const __bf16* __restrict__ xe, const int* __restrict__ x,
    const __bf16* __restrict__ Wf_p, const __bf16* __restrict__ Uf_p,
    const __bf16* __restrict__ Wb_p, const __bf16* __restrict__ Ub_p,
    const float* __restrict__ bf_in, const float* __restrict__ bf_rec,
    const float* __restrict__ bb_in, const float* __restrict__ bb_rec,
    const __bf16* __restrict__ hf_prev, __bf16* __restrict__ hf_next,
    const __bf16* __restrict__ hb_prev, __bf16* __restrict__ hb_next,
    float* __restrict__ out) {
  const int dir  = blockIdx.x >> 4;
  const int wg   = blockIdx.x & 15;
  const int wave = threadIdx.x >> 5;
  const int lane = threadIdx.x & 31;
  const int t    = dir ? (SEQ - 1 - s) : s;

  const __bf16* Wpk = dir ? Wb_p : Wf_p;
  const __bf16* Upk = dir ? Ub_p : Uf_p;
  const float*  bin = dir ? bb_in  : bf_in;
  const float*  brc = dir ? bb_rec : bf_rec;
  const __bf16* hprev = dir ? hb_prev : hf_prev;
  __bf16*       hnext = dir ? hb_next : hf_next;

  const int mt   = wave & 3;
  const int nt0  = (wave >> 2) * 2;
  const int arow = mt * 16 + (lane & 15);          // batch row for A frags
  const int koff = (lane & 16) ? 8 : 0;
  const int NTW  = G3 >> 4;                        // 192 n-tiles in W/U

  v8f accz[2] = {}, accr[2] = {}, accxh[2] = {}, accrh[2] = {};

  // --- input projection: K = EMB (8 k-tiles), A = xe[:, t, :] ---
  const __bf16* xebase = xe + ((size_t)arow * SEQ + t) * EMB;
  for (int kt = 0; kt < EMB / 32; ++kt) {
    v16bf a = load_a(xebase + kt * 32 + koff);
#pragma unroll
    for (int j = 0; j < 2; ++j) {
      int ntg = wg * 4 + nt0 + j;
      v16bf bz = load_b(Wpk + (((size_t)kt * NTW + ntg      ) * 32 + lane) * 16);
      v16bf br = load_b(Wpk + (((size_t)kt * NTW + ntg + 64 ) * 32 + lane) * 16);
      v16bf bh = load_b(Wpk + (((size_t)kt * NTW + ntg + 128) * 32 + lane) * 16);
      accz[j]  = wmma_bf16(a, bz, accz[j]);
      accr[j]  = wmma_bf16(a, br, accr[j]);
      accxh[j] = wmma_bf16(a, bh, accxh[j]);
    }
  }
  // --- recurrent: K = UNITS (32 k-tiles), A = h_prev ---
  const __bf16* hbase = hprev + (size_t)arow * UNITS;
  for (int kt = 0; kt < UNITS / 32; ++kt) {
    v16bf a = load_a(hbase + kt * 32 + koff);
#pragma unroll
    for (int j = 0; j < 2; ++j) {
      int ntg = wg * 4 + nt0 + j;
      v16bf bz = load_b(Upk + (((size_t)kt * NTW + ntg      ) * 32 + lane) * 16);
      v16bf br = load_b(Upk + (((size_t)kt * NTW + ntg + 64 ) * 32 + lane) * 16);
      v16bf bh = load_b(Upk + (((size_t)kt * NTW + ntg + 128) * 32 + lane) * 16);
      accz[j]  = wmma_bf16(a, bz, accz[j]);
      accr[j]  = wmma_bf16(a, br, accr[j]);
      accrh[j] = wmma_bf16(a, bh, accrh[j]);
    }
  }

  // --- gates + state update (C layout: VGPR v -> M = v + 8*(lane>=16)) ---
  const int crow = mt * 16 + ((lane & 16) ? 8 : 0);
#pragma unroll
  for (int j = 0; j < 2; ++j) {
    int j0  = wg * 64 + (nt0 + j) * 16 + (lane & 15);   // unit index
    float bz  = bin[j0]             + brc[j0];
    float brr = bin[UNITS + j0]     + brc[UNITS + j0];
    float bxh = bin[2 * UNITS + j0];
    float brh = brc[2 * UNITS + j0];
#pragma unroll
    for (int v = 0; v < 8; ++v) {
      int row  = crow + v;
      float hp = (float)hprev[(size_t)row * UNITS + j0];
      float z  = sigmoidf_(accz[j][v] + bz);
      float r  = sigmoidf_(accr[j][v] + brr);
      float hh = tanhf(accxh[j][v] + bxh + r * (accrh[j][v] + brh));
      float hn = z * hp + (1.0f - z) * hh;
      if (x[row * SEQ + t] == 0) hn = hp;              // mask
      hnext[(size_t)row * UNITS + j0] = (__bf16)hn;
      out[((size_t)row * SEQ + t) * (2 * UNITS) + dir * UNITS + j0] = hn;
    }
  }
}

// ---- final state: tanh([hf|hb] @ Wp + bp) -----------------------------------
__global__ __launch_bounds__(256) void k_state(
    const __bf16* __restrict__ hf, const __bf16* __restrict__ hb,
    const __bf16* __restrict__ Wp_p, const float* __restrict__ bp,
    float* __restrict__ state) {
  const int wg   = blockIdx.x;           // 16
  const int wave = threadIdx.x >> 5;
  const int lane = threadIdx.x & 31;
  const int mt   = wave & 3;
  const int nt0  = (wave >> 2) * 2;
  const int arow = mt * 16 + (lane & 15);
  const int koff = (lane & 16) ? 8 : 0;
  const int NT   = UNITS >> 4;           // 64

  v8f acc[2] = {};
  for (int kt = 0; kt < 64; ++kt) {      // K = 2048
    const __bf16* hsrc = (kt < 32) ? hf : hb;
    int kc = (kt & 31) * 32;
    v16bf a = load_a(hsrc + (size_t)arow * UNITS + kc + koff);
#pragma unroll
    for (int j = 0; j < 2; ++j) {
      int ntg = wg * 4 + nt0 + j;
      v16bf b = load_b(Wp_p + (((size_t)kt * NT + ntg) * 32 + lane) * 16);
      acc[j] = wmma_bf16(a, b, acc[j]);
    }
  }
  const int crow = mt * 16 + ((lane & 16) ? 8 : 0);
#pragma unroll
  for (int j = 0; j < 2; ++j) {
    int j0  = wg * 64 + (nt0 + j) * 16 + (lane & 15);
    float b = bp[j0];
#pragma unroll
    for (int v = 0; v < 8; ++v)
      state[(size_t)(crow + v) * UNITS + j0] = tanhf(acc[j][v] + b);
  }
}

// ---- host -------------------------------------------------------------------
extern "C" void kernel_launch(void* const* d_in, const int* in_sizes, int n_in,
                              void* d_out, int out_size, void* d_ws, size_t ws_size,
                              hipStream_t stream) {
  const int*   x      = (const int*)  d_in[0];
  const float* hidden = (const float*)d_in[1];
  const float* emb    = (const float*)d_in[2];
  const float* Wf     = (const float*)d_in[3];
  const float* Uf     = (const float*)d_in[4];
  const float* bf_in  = (const float*)d_in[5];
  const float* bf_rec = (const float*)d_in[6];
  const float* Wb     = (const float*)d_in[7];
  const float* Ub     = (const float*)d_in[8];
  const float* bb_in  = (const float*)d_in[9];
  const float* bb_rec = (const float*)d_in[10];
  const float* Wp     = (const float*)d_in[11];
  const float* bp     = (const float*)d_in[12];

  float* out   = (float*)d_out;
  float* state = out + (size_t)BATCH * SEQ * 2 * UNITS;

  char* ws = (char*)d_ws;
  size_t off = 0;
  auto alloc = [&](size_t bytes) -> char* {
    char* p = ws + off;
    off += (bytes + 255) & ~(size_t)255;
    return p;
  };
  __bf16* xe   = (__bf16*)alloc((size_t)BATCH * SEQ * EMB * 2);
  __bf16* Wf_p = (__bf16*)alloc((size_t)EMB * G3 * 2);
  __bf16* Wb_p = (__bf16*)alloc((size_t)EMB * G3 * 2);
  __bf16* Uf_p = (__bf16*)alloc((size_t)UNITS * G3 * 2);
  __bf16* Ub_p = (__bf16*)alloc((size_t)UNITS * G3 * 2);
  __bf16* Wp_p = (__bf16*)alloc((size_t)2 * UNITS * UNITS * 2);
  __bf16* hf0  = (__bf16*)alloc((size_t)BATCH * UNITS * 2);
  __bf16* hf1  = (__bf16*)alloc((size_t)BATCH * UNITS * 2);
  __bf16* hb0  = (__bf16*)alloc((size_t)BATCH * UNITS * 2);
  __bf16* hb1  = (__bf16*)alloc((size_t)BATCH * UNITS * 2);

  k_embed<<<(BATCH * SEQ * EMB / 4) / 256, 256, 0, stream>>>(x, emb, xe);
  k_pack<<<(EMB * G3 + 255) / 256, 256, 0, stream>>>(Wf, Wf_p, EMB, G3);
  k_pack<<<(EMB * G3 + 255) / 256, 256, 0, stream>>>(Wb, Wb_p, EMB, G3);
  k_pack<<<(UNITS * G3 + 255) / 256, 256, 0, stream>>>(Uf, Uf_p, UNITS, G3);
  k_pack<<<(UNITS * G3 + 255) / 256, 256, 0, stream>>>(Ub, Ub_p, UNITS, G3);
  k_pack<<<(2 * UNITS * UNITS + 255) / 256, 256, 0, stream>>>(Wp, Wp_p, 2 * UNITS, UNITS);
  k_h0<<<(BATCH * UNITS) / 256, 256, 0, stream>>>(hidden, hf0, hb0);

  for (int s = 0; s < SEQ; ++s) {
    const __bf16* hf_prev = (s & 1) ? hf1 : hf0;
    __bf16*       hf_next = (s & 1) ? hf0 : hf1;
    const __bf16* hb_prev = (s & 1) ? hb1 : hb0;
    __bf16*       hb_next = (s & 1) ? hb0 : hb1;
    k_gru_step<<<32, 256, 0, stream>>>(s, xe, x, Wf_p, Uf_p, Wb_p, Ub_p,
                                       bf_in, bf_rec, bb_in, bb_rec,
                                       hf_prev, hf_next, hb_prev, hb_next, out);
  }
  // SEQ = 256 (even): last step (s = 255, odd) wrote hf0/hb0
  k_state<<<16, 256, 0, stream>>>(hf0, hb0, Wp_p, bp, state);
}
